// PointTrans_Layer_23673859735698
// MI455X (gfx1250) — compile-verified, matrix operand order checked by reading, not run on hardware
//
#include <hip/hip_runtime.h>
#include <hip/hip_bf16.h>
#include <stdint.h>

typedef __bf16 bf16_t;
typedef __attribute__((ext_vector_type(16))) __bf16 v16bf;
typedef __attribute__((ext_vector_type(8)))  __bf16 v8bf;
typedef __attribute__((ext_vector_type(8)))  float   v8f;

#define TPB 256

__device__ __forceinline__ v8f wmma_bf16(v16bf a, v16bf b, v8f c) {
  return __builtin_amdgcn_wmma_f32_16x16x32_bf16(false, a, false, b, (short)0, c,
                                                 false, false);
}

__device__ __forceinline__ void load8(const float* __restrict__ p, float* t) {
  float4 u = *(const float4*)p;
  float4 v = *(const float4*)(p + 4);
  t[0] = u.x; t[1] = u.y; t[2] = u.z; t[3] = u.w;
  t[4] = v.x; t[5] = v.y; t[6] = v.z; t[7] = v.w;
}

__device__ __forceinline__ void store8(float* __restrict__ p, const float* t) {
  *(float4*)p       = make_float4(t[0], t[1], t[2], t[3]);
  *(float4*)(p + 4) = make_float4(t[4], t[5], t[6], t[7]);
}

__device__ __forceinline__ void pack8(v16bf& a, int base, const float* t) {
#pragma unroll
  for (int i = 0; i < 8; ++i) a[base + i] = (bf16_t)t[i];
}

// B fragment for X @ W^T, W (64x64) bf16 in LDS; per-lane K chunks {k0+8h, k0+16+8h}.
__device__ __forceinline__ v16bf load_wfrag_lds(const bf16_t* sW, int n0, int k0,
                                                int nn, int h) {
  const bf16_t* base = sW + (n0 + nn) * 64 + k0 + 8 * h;
  v8bf lo = *(const v8bf*)base;
  v8bf hi = *(const v8bf*)(base + 16);
  v16bf b;
#pragma unroll
  for (int i = 0; i < 8; ++i) { b[i] = lo[i]; b[8 + i] = hi[i]; }
  return b;
}

// ---- CDNA5 async global->LDS (GLOBAL_LOAD_ASYNC_TO_LDS_B128, ASYNCcnt) ----
__device__ __forceinline__ void async_copy_b128(void* lds_dst, const void* gsrc) {
  // low 32 bits of a generic shared pointer == LDS byte offset (aperture layout)
  unsigned lo = (unsigned)(uintptr_t)lds_dst;
  asm volatile("global_load_async_to_lds_b128 %0, %1, off" ::"v"(lo), "v"(gsrc)
               : "memory");
}
__device__ __forceinline__ void async_wait0() {
  asm volatile("s_wait_asynccnt 0" ::: "memory");
}
__device__ __forceinline__ void fill_w_lds_issue(bf16_t* sW, const bf16_t* gW,
                                                 int chunks16B) {
  for (int i = threadIdx.x; i < chunks16B; i += blockDim.x)
    async_copy_b128(sW + i * 8, gW + i * 8);
}

// ------- one-time f32 -> bf16 conversion of the 7 64x64 weight matrices -------
struct WPtrs { const float* w[7]; };
__global__ void convert_weights_kernel(WPtrs wp, bf16_t* __restrict__ out) {
  int i = blockIdx.x * blockDim.x + threadIdx.x;
  if (i < 7 * 4096) out[i] = (bf16_t)wp.w[i >> 12][i & 4095];
}

// ---------------- node projections: val/a_src/a_dst = x @ W^T ----------------
__global__ void __launch_bounds__(TPB)
node_proj_kernel(const float* __restrict__ x, const bf16_t* __restrict__ wbf,
                 float* __restrict__ val, float* __restrict__ asrc,
                 float* __restrict__ adst, int n) {
  __shared__ bf16_t sW[3 * 4096];
  fill_w_lds_issue(sW, wbf, 3 * 512);
  async_wait0();
  __syncthreads();
  int lane = threadIdx.x & 31;
  long wave = ((long)blockIdx.x * blockDim.x + threadIdx.x) >> 5;
  long row0 = wave * 16;
  if (row0 >= n) return;
  int m = lane & 15, h = (lane >> 4) & 1, nn = m;
  long r = row0 + m;
  if (r >= n) r = n - 1;
  const float* row = x + r * 64;
  v16bf a0, a1;
#pragma unroll
  for (int ci = 0; ci < 4; ++ci) {
    int kk = 16 * ci + 8 * h;
    float t[8];
    load8(row + kk, t);
    pack8(ci >= 2 ? a1 : a0, (ci & 1) * 8, t);
  }
  bool full = (row0 + 16 <= n);
  float* outs[3] = {val, asrc, adst};
#pragma unroll
  for (int wi = 0; wi < 3; ++wi) {
    const bf16_t* sWm = sW + wi * 4096;
    float* ob = outs[wi] + (row0 + 8 * h) * 64 + nn;
#pragma unroll
    for (int n0 = 0; n0 < 64; n0 += 16) {
      v8f c = {};
      c = wmma_bf16(a0, load_wfrag_lds(sWm, n0, 0, nn, h), c);
      c = wmma_bf16(a1, load_wfrag_lds(sWm, n0, 32, nn, h), c);
      if (full) {
#pragma unroll
        for (int rr = 0; rr < 8; ++rr) ob[n0 + rr * 64] = c[rr];
      } else {
#pragma unroll
        for (int rr = 0; rr < 8; ++rr)
          if (row0 + 8 * h + rr < n) ob[n0 + rr * 64] = c[rr];
      }
    }
  }
}

// -------- moments of t = pos_d - pos_s: S (3) and upper-tri Q (6) ----------
__global__ void __launch_bounds__(TPB)
moments_kernel(const float* __restrict__ pos, const int* __restrict__ ei, long E,
               long M, float* __restrict__ mom) {
  __shared__ float sred[9];
  if (threadIdx.x < 9) sred[threadIdx.x] = 0.f;
  __syncthreads();
  long e = (long)blockIdx.x * blockDim.x + threadIdx.x;
  float v[9];
#pragma unroll
  for (int j = 0; j < 9; ++j) v[j] = 0.f;
  if (e < M) {
    long s, d;
    if (e < E) { s = ei[e]; d = ei[E + e]; } else { s = e - E; d = s; }
    float dx = pos[d * 3 + 0] - pos[s * 3 + 0];
    float dy = pos[d * 3 + 1] - pos[s * 3 + 1];
    float dz = pos[d * 3 + 2] - pos[s * 3 + 2];
    v[0] = dx; v[1] = dy; v[2] = dz;
    v[3] = dx * dx; v[4] = dx * dy; v[5] = dx * dz;
    v[6] = dy * dy; v[7] = dy * dz; v[8] = dz * dz;
  }
#pragma unroll
  for (int j = 0; j < 9; ++j) {
    float a = v[j];
#pragma unroll
    for (int off = 16; off; off >>= 1) a += __shfl_xor(a, off, 32);
    if ((threadIdx.x & 31) == 0) atomicAdd(&sred[j], a);
  }
  __syncthreads();
  if (threadIdx.x < 9) atomicAdd(&mom[threadIdx.x], sred[threadIdx.x]);
}

// ------ BN1 affine from moments: h = w.t + b is linear in t ------------------
__global__ void bn1_finalize_kernel(const float* __restrict__ mom,
                                    const float* __restrict__ w1,
                                    const float* __restrict__ b1,
                                    const float* __restrict__ g,
                                    const float* __restrict__ beta, float invM,
                                    float* __restrict__ sab) {
  int c = threadIdx.x;  // 64 threads
  float wx = w1[c * 3], wy = w1[c * 3 + 1], wz = w1[c * 3 + 2], b = b1[c];
  float Sx = mom[0] * invM, Sy = mom[1] * invM, Sz = mom[2] * invM;
  float Qxx = mom[3] * invM, Qxy = mom[4] * invM, Qxz = mom[5] * invM;
  float Qyy = mom[6] * invM, Qyz = mom[7] * invM, Qzz = mom[8] * invM;
  float wS = wx * Sx + wy * Sy + wz * Sz;
  float mean = wS + b;
  float quad = wx * wx * Qxx + wy * wy * Qyy + wz * wz * Qzz +
               2.f * (wx * wy * Qxy + wx * wz * Qxz + wy * wz * Qyz);
  float var = (quad + 2.f * b * wS + b * b) - mean * mean;
  float is = rsqrtf(var + 1e-5f);
  float sA = is * g[c];
  sab[c] = sA;
  sab[64 + c] = beta[c] - mean * sA;
}

__global__ void bn_finalize_kernel(const float* __restrict__ stats,
                                   const float* __restrict__ g,
                                   const float* __restrict__ beta, float invM,
                                   float* __restrict__ sab) {
  int c = threadIdx.x;
  float mean = stats[c] * invM;
  float var = stats[64 + c] * invM - mean * mean;
  float is = rsqrtf(var + 1e-5f);
  float sA = is * g[c];
  sab[c] = sA;
  sab[64 + c] = beta[c] - mean * sA;
}

// ---- stage 1+2 fused: hbuf = relu(bn1(t.W1^T + b1)) @ W2^T + b2 ------------
__global__ void __launch_bounds__(TPB)
edge_gemm1_kernel(const float* __restrict__ pos, const int* __restrict__ ei,
                  long E, long M, const float* __restrict__ w1,
                  const float* __restrict__ b1, const float* __restrict__ sab1,
                  const bf16_t* __restrict__ gW, const float* __restrict__ bias,
                  float* __restrict__ hbuf, float* __restrict__ stats_out) {
  __shared__ bf16_t sW[4096];
  __shared__ float sw1[192], sb1[64], ssab[128], sred[128];
  fill_w_lds_issue(sW, gW, 512);
  for (int i = threadIdx.x; i < 192; i += blockDim.x) sw1[i] = w1[i];
  for (int i = threadIdx.x; i < 64; i += blockDim.x) sb1[i] = b1[i];
  for (int i = threadIdx.x; i < 128; i += blockDim.x) {
    ssab[i] = sab1[i];
    sred[i] = 0.f;
  }
  async_wait0();
  __syncthreads();
  int lane = threadIdx.x & 31;
  long wave = ((long)blockIdx.x * blockDim.x + threadIdx.x) >> 5;
  long row0 = wave * 16;
  if (row0 < M) {
    int m = lane & 15, h = (lane >> 4) & 1, nn = m;
    long r = row0 + m;
    if (r >= M) r = M - 1;
    long s, d;
    if (r < E) { s = ei[r]; d = ei[E + r]; } else { s = r - E; d = s; }
    float tx = pos[d * 3 + 0] - pos[s * 3 + 0];
    float ty = pos[d * 3 + 1] - pos[s * 3 + 1];
    float tz = pos[d * 3 + 2] - pos[s * 3 + 2];
    v16bf a0, a1;
#pragma unroll
    for (int ci = 0; ci < 4; ++ci) {
      int kk = 16 * ci + 8 * h;
      float u[8];
#pragma unroll
      for (int i = 0; i < 8; ++i) {
        int k = kk + i;
        float hv = fmaf(tx, sw1[k * 3],
                   fmaf(ty, sw1[k * 3 + 1], fmaf(tz, sw1[k * 3 + 2], sb1[k])));
        float vv = fmaf(hv, ssab[k], ssab[64 + k]);
        u[i] = vv > 0.f ? vv : 0.f;
      }
      pack8(ci >= 2 ? a1 : a0, (ci & 1) * 8, u);
    }
    bool full = (row0 + 16 <= M);
    float* ob = hbuf + (row0 + 8 * h) * 64 + nn;
#pragma unroll
    for (int n0 = 0; n0 < 64; n0 += 16) {
      v8f c = {};
      c = wmma_bf16(a0, load_wfrag_lds(sW, n0, 0, nn, h), c);
      c = wmma_bf16(a1, load_wfrag_lds(sW, n0, 32, nn, h), c);
      float bv = bias[n0 + nn];
      float sacc = 0.f, qacc = 0.f;
      if (full) {
#pragma unroll
        for (int rr = 0; rr < 8; ++rr) {
          float v = c[rr] + bv;
          ob[n0 + rr * 64] = v;
          sacc += v; qacc += v * v;
        }
      } else {
#pragma unroll
        for (int rr = 0; rr < 8; ++rr)
          if (row0 + 8 * h + rr < M) {
            float v = c[rr] + bv;
            ob[n0 + rr * 64] = v;
            sacc += v; qacc += v * v;
          }
      }
      sacc += __shfl_xor(sacc, 16, 32);
      qacc += __shfl_xor(qacc, 16, 32);
      if (h == 0) {
        atomicAdd(&sred[n0 + nn], sacc);
        atomicAdd(&sred[64 + n0 + nn], qacc);
      }
    }
  }
  __syncthreads();
  for (int i = threadIdx.x; i < 128; i += blockDim.x)
    atomicAdd(&stats_out[i], sred[i]);
}

// -------- generic in-place stage: hbuf <- relu(bn(hbuf)) @ W^T + bias --------
__global__ void __launch_bounds__(TPB)
edge_gemm_kernel(float* __restrict__ hbuf, long M, const float* __restrict__ sab,
                 const bf16_t* __restrict__ gW, const float* __restrict__ bias,
                 float* __restrict__ stats_out) {
  __shared__ bf16_t sW[4096];
  __shared__ float ssab[128], sred[128];
  fill_w_lds_issue(sW, gW, 512);
  for (int i = threadIdx.x; i < 128; i += blockDim.x) {
    ssab[i] = sab[i];
    sred[i] = 0.f;
  }
  async_wait0();
  __syncthreads();
  int lane = threadIdx.x & 31;
  long wave = ((long)blockIdx.x * blockDim.x + threadIdx.x) >> 5;
  long row0 = wave * 16;
  if (row0 < M) {
    int m = lane & 15, h = (lane >> 4) & 1, nn = m;
    long r = row0 + m;
    if (r >= M) r = M - 1;
    const float* row = hbuf + r * 64;
    v16bf a0, a1;
#pragma unroll
    for (int ci = 0; ci < 4; ++ci) {
      int kk = 16 * ci + 8 * h;
      float t[8], u[8];
      load8(row + kk, t);
#pragma unroll
      for (int i = 0; i < 8; ++i) {
        float vv = fmaf(t[i], ssab[kk + i], ssab[64 + kk + i]);
        u[i] = vv > 0.f ? vv : 0.f;
      }
      pack8(ci >= 2 ? a1 : a0, (ci & 1) * 8, u);
    }
    bool full = (row0 + 16 <= M);
    float* ob = hbuf + (row0 + 8 * h) * 64 + nn;
#pragma unroll
    for (int n0 = 0; n0 < 64; n0 += 16) {
      v8f c = {};
      c = wmma_bf16(a0, load_wfrag_lds(sW, n0, 0, nn, h), c);
      c = wmma_bf16(a1, load_wfrag_lds(sW, n0, 32, nn, h), c);
      float bv = bias[n0 + nn];
      float sacc = 0.f, qacc = 0.f;
      if (full) {
#pragma unroll
        for (int rr = 0; rr < 8; ++rr) {
          float v = c[rr] + bv;
          ob[n0 + rr * 64] = v;
          sacc += v; qacc += v * v;
        }
      } else {
#pragma unroll
        for (int rr = 0; rr < 8; ++rr)
          if (row0 + 8 * h + rr < M) {
            float v = c[rr] + bv;
            ob[n0 + rr * 64] = v;
            sacc += v; qacc += v * v;
          }
      }
      sacc += __shfl_xor(sacc, 16, 32);
      qacc += __shfl_xor(qacc, 16, 32);
      if (h == 0) {
        atomicAdd(&sred[n0 + nn], sacc);
        atomicAdd(&sred[64 + n0 + nn], qacc);
      }
    }
  }
  __syncthreads();
  for (int i = threadIdx.x; i < 128; i += blockDim.x)
    atomicAdd(&stats_out[i], sred[i]);
}

// -- delta = relu(bn2(hbuf)) -> dbuf; hbuf <- (a_dst[d]-a_src[s]+delta)@W^T+b --
__global__ void __launch_bounds__(TPB)
edge_attn_in_kernel(float* __restrict__ hbuf, float* __restrict__ dbuf, long M,
                    long E, const int* __restrict__ ei,
                    const float* __restrict__ sab2,
                    const float* __restrict__ asrc,
                    const float* __restrict__ adst,
                    const bf16_t* __restrict__ gW,
                    const float* __restrict__ bias,
                    float* __restrict__ stats_out) {
  __shared__ bf16_t sW[4096];
  __shared__ float ssab[128], sred[128];
  fill_w_lds_issue(sW, gW, 512);
  for (int i = threadIdx.x; i < 128; i += blockDim.x) {
    ssab[i] = sab2[i];
    sred[i] = 0.f;
  }
  async_wait0();
  __syncthreads();
  int lane = threadIdx.x & 31;
  long wave = ((long)blockIdx.x * blockDim.x + threadIdx.x) >> 5;
  long row0 = wave * 16;
  if (row0 < M) {
    int m = lane & 15, h = (lane >> 4) & 1, nn = m;
    long r = row0 + m;
    if (r >= M) r = M - 1;
    long s, d;
    if (r < E) { s = ei[r]; d = ei[E + r]; } else { s = r - E; d = s; }
    const float* hrow = hbuf + r * 64;
    const float* srow = asrc + s * 64;
    const float* drow = adst + d * 64;
    float* dout = dbuf + r * 64;
    v16bf a0, a1;
#pragma unroll
    for (int ci = 0; ci < 4; ++ci) {
      int kk = 16 * ci + 8 * h;
      float t[8], ts[8], td[8], dv[8], u[8];
      load8(hrow + kk, t);
      load8(srow + kk, ts);
      load8(drow + kk, td);
#pragma unroll
      for (int i = 0; i < 8; ++i) {
        float vv = fmaf(t[i], ssab[kk + i], ssab[64 + kk + i]);
        dv[i] = vv > 0.f ? vv : 0.f;
        u[i] = td[i] - ts[i] + dv[i];
      }
      store8(dout + kk, dv);  // lanes m and m+16 cover complementary columns
      pack8(ci >= 2 ? a1 : a0, (ci & 1) * 8, u);
    }
    bool full = (row0 + 16 <= M);
    float* ob = hbuf + (row0 + 8 * h) * 64 + nn;
#pragma unroll
    for (int n0 = 0; n0 < 64; n0 += 16) {
      v8f c = {};
      c = wmma_bf16(a0, load_wfrag_lds(sW, n0, 0, nn, h), c);
      c = wmma_bf16(a1, load_wfrag_lds(sW, n0, 32, nn, h), c);
      float bv = bias[n0 + nn];
      float sacc = 0.f, qacc = 0.f;
      if (full) {
#pragma unroll
        for (int rr = 0; rr < 8; ++rr) {
          float v = c[rr] + bv;
          ob[n0 + rr * 64] = v;
          sacc += v; qacc += v * v;
        }
      } else {
#pragma unroll
        for (int rr = 0; rr < 8; ++rr)
          if (row0 + 8 * h + rr < M) {
            float v = c[rr] + bv;
            ob[n0 + rr * 64] = v;
            sacc += v; qacc += v * v;
          }
      }
      sacc += __shfl_xor(sacc, 16, 32);
      qacc += __shfl_xor(qacc, 16, 32);
      if (h == 0) {
        atomicAdd(&sred[n0 + nn], sacc);
        atomicAdd(&sred[64 + n0 + nn], qacc);
      }
    }
  }
  __syncthreads();
  for (int i = threadIdx.x; i < 128; i += blockDim.x)
    atomicAdd(&stats_out[i], sred[i]);
}

// ---------- segment softmax, 4 channels per thread (b128 traffic) -----------
__global__ void __launch_bounds__(TPB)
softmax_max_kernel(float* __restrict__ hbuf, long M, long E,
                   const int* __restrict__ ei, const float* __restrict__ sab4,
                   unsigned int* __restrict__ segmax) {
  long i = (long)blockIdx.x * blockDim.x + threadIdx.x;
  if (i >= M * 16) return;
  long e = i >> 4;
  int c0 = (int)(i & 15) * 4;
  long d = (e < E) ? (long)ei[E + e] : (e - E);
  float4 v = *(float4*)(hbuf + e * 64 + c0);
  float4 sA = *(const float4*)(sab4 + c0);
  float4 sB = *(const float4*)(sab4 + 64 + c0);
  float r0 = fmaxf(fmaf(v.x, sA.x, sB.x), 0.f);
  float r1 = fmaxf(fmaf(v.y, sA.y, sB.y), 0.f);
  float r2 = fmaxf(fmaf(v.z, sA.z, sB.z), 0.f);
  float r3 = fmaxf(fmaf(v.w, sA.w, sB.w), 0.f);
  *(float4*)(hbuf + e * 64 + c0) = make_float4(r0, r1, r2, r3);
  unsigned int* sm = segmax + d * 64 + c0;  // uint-max == float-max for v >= 0
  atomicMax(sm + 0, __float_as_uint(r0));
  atomicMax(sm + 1, __float_as_uint(r1));
  atomicMax(sm + 2, __float_as_uint(r2));
  atomicMax(sm + 3, __float_as_uint(r3));
}

__global__ void __launch_bounds__(TPB)
softmax_exp_kernel(float* __restrict__ hbuf, long M, long E,
                   const int* __restrict__ ei,
                   const unsigned int* __restrict__ segmax,
                   float* __restrict__ segsum) {
  long i = (long)blockIdx.x * blockDim.x + threadIdx.x;
  if (i >= M * 16) return;
  long e = i >> 4;
  int c0 = (int)(i & 15) * 4;
  long d = (e < E) ? (long)ei[E + e] : (e - E);
  float4 v = *(float4*)(hbuf + e * 64 + c0);
  const unsigned int* sm = segmax + d * 64 + c0;
  float e0 = __expf(v.x - __uint_as_float(sm[0]));
  float e1 = __expf(v.y - __uint_as_float(sm[1]));
  float e2 = __expf(v.z - __uint_as_float(sm[2]));
  float e3 = __expf(v.w - __uint_as_float(sm[3]));
  *(float4*)(hbuf + e * 64 + c0) = make_float4(e0, e1, e2, e3);
  float* ss = segsum + d * 64 + c0;
  atomicAdd(ss + 0, e0);
  atomicAdd(ss + 1, e1);
  atomicAdd(ss + 2, e2);
  atomicAdd(ss + 3, e3);
}

__global__ void __launch_bounds__(TPB)
scatter_kernel(const float* __restrict__ hbuf, const float* __restrict__ dbuf,
               long M, long E, const int* __restrict__ ei,
               const float* __restrict__ segsum, const float* __restrict__ val,
               float* __restrict__ outacc) {
  long i = (long)blockIdx.x * blockDim.x + threadIdx.x;
  if (i >= M * 16) return;
  long e = i >> 4;
  int c0 = (int)(i & 15) * 4;
  long s, d;
  if (e < E) { s = ei[e]; d = ei[E + e]; } else { s = e - E; d = s; }
  float4 a = *(const float4*)(hbuf + e * 64 + c0);
  float4 dl = *(const float4*)(dbuf + e * 64 + c0);
  float4 vl = *(const float4*)(val + s * 64 + c0);
  const float* ss = segsum + d * 64 + c0;
  float* oa = outacc + d * 64 + c0;
  atomicAdd(oa + 0, a.x / (ss[0] + 1e-16f) * (vl.x + dl.x));
  atomicAdd(oa + 1, a.y / (ss[1] + 1e-16f) * (vl.y + dl.y));
  atomicAdd(oa + 2, a.z / (ss[2] + 1e-16f) * (vl.z + dl.z));
  atomicAdd(oa + 3, a.w / (ss[3] + 1e-16f) * (vl.w + dl.w));
}

// ---------------- final: out = outacc @ up_w^T + up_b + x -------------------
__global__ void __launch_bounds__(TPB)
final_kernel(const float* __restrict__ acc, const float* __restrict__ x,
             const bf16_t* __restrict__ gW, const float* __restrict__ bias,
             float* __restrict__ out, int n) {
  __shared__ bf16_t sW[4096];
  fill_w_lds_issue(sW, gW, 512);
  async_wait0();
  __syncthreads();
  int lane = threadIdx.x & 31;
  long wave = ((long)blockIdx.x * blockDim.x + threadIdx.x) >> 5;
  long row0 = wave * 16;
  if (row0 >= n) return;
  int m = lane & 15, h = (lane >> 4) & 1, nn = m;
  long r = row0 + m;
  if (r >= n) r = n - 1;
  const float* row = acc + r * 64;
  v16bf a0, a1;
#pragma unroll
  for (int ci = 0; ci < 4; ++ci) {
    int kk = 16 * ci + 8 * h;
    float t[8];
    load8(row + kk, t);
    pack8(ci >= 2 ? a1 : a0, (ci & 1) * 8, t);
  }
  bool full = (row0 + 16 <= n);
  long obase = (row0 + 8 * h) * 64 + nn;
#pragma unroll
  for (int n0 = 0; n0 < 64; n0 += 16) {
    v8f c = {};
    c = wmma_bf16(a0, load_wfrag_lds(sW, n0, 0, nn, h), c);
    c = wmma_bf16(a1, load_wfrag_lds(sW, n0, 32, nn, h), c);
    float bv = bias[n0 + nn];
    if (full) {
#pragma unroll
      for (int rr = 0; rr < 8; ++rr)
        out[obase + n0 + rr * 64] = c[rr] + bv + x[obase + n0 + rr * 64];
    } else {
#pragma unroll
      for (int rr = 0; rr < 8; ++rr)
        if (row0 + 8 * h + rr < n)
          out[obase + n0 + rr * 64] = c[rr] + bv + x[obase + n0 + rr * 64];
    }
  }
}

extern "C" void kernel_launch(void* const* d_in, const int* in_sizes, int n_in,
                              void* d_out, int out_size, void* d_ws, size_t ws_size,
                              hipStream_t stream) {
  const float* x        = (const float*)d_in[0];
  const float* pos      = (const float*)d_in[1];
  const int*   ei       = (const int*)d_in[2];
  const float* lin_w    = (const float*)d_in[3];
  const float* lin_src_w= (const float*)d_in[4];
  const float* lin_dst_w= (const float*)d_in[5];
  const float* pos_w1   = (const float*)d_in[6];
  const float* pos_b1   = (const float*)d_in[7];
  const float* pos_g1   = (const float*)d_in[8];
  const float* pos_be1  = (const float*)d_in[9];
  const float* pos_w2   = (const float*)d_in[10];
  const float* pos_b2   = (const float*)d_in[11];
  const float* pos_g2   = (const float*)d_in[12];
  const float* pos_be2  = (const float*)d_in[13];
  const float* attn_w1  = (const float*)d_in[14];
  const float* attn_b1  = (const float*)d_in[15];
  const float* attn_g1  = (const float*)d_in[16];
  const float* attn_be1 = (const float*)d_in[17];
  const float* attn_w2  = (const float*)d_in[18];
  const float* attn_b2  = (const float*)d_in[19];
  const float* attn_g2  = (const float*)d_in[20];
  const float* attn_be2 = (const float*)d_in[21];
  const float* up_w     = (const float*)d_in[22];
  const float* up_b     = (const float*)d_in[23];

  const long N = in_sizes[0] / 64;
  const long E = in_sizes[2] / 2;
  const long M = N + E;

  char* p = (char*)d_ws;
  auto carve = [&](size_t bytes) -> char* {
    char* q = p;
    p += (bytes + 255) & ~(size_t)255;
    return q;
  };
  bf16_t* wbf  = (bf16_t*)carve(7 * 4096 * sizeof(bf16_t));
  float* asrc  = (float*)carve((size_t)N * 64 * 4);
  float* adst  = (float*)carve((size_t)N * 64 * 4);
  float* val   = (float*)carve((size_t)N * 64 * 4);
  float* hbuf  = (float*)carve((size_t)M * 64 * 4);
  float* dbuf  = (float*)carve((size_t)M * 64 * 4);
  char* zbase = p;  // contiguous zero-initialized region
  unsigned int* segmax = (unsigned int*)carve((size_t)N * 64 * 4);
  float* segsum = (float*)carve((size_t)N * 64 * 4);
  float* outacc = (float*)carve((size_t)N * 64 * 4);
  float* mstats = (float*)carve((16 + 3 * 128) * 4);  // mom[9] | stats2,3,4
  size_t zbytes = (size_t)(p - zbase);
  float* sab = (float*)carve(4 * 128 * 4);

  float* mom    = mstats;
  float* stats2 = mstats + 16;
  float* stats3 = mstats + 16 + 128;
  float* stats4 = mstats + 16 + 256;

  hipMemsetAsync(zbase, 0, zbytes, stream);

  const float invM = 1.0f / (float)M;
  const unsigned nodeBlocks = (unsigned)((((N + 15) / 16) * 32 + TPB - 1) / TPB);
  const unsigned edgeBlocks = (unsigned)((M + TPB - 1) / TPB);
  const unsigned gemmBlocks = (unsigned)((((M + 15) / 16) * 32 + TPB - 1) / TPB);
  const unsigned elemBlocks = (unsigned)((M * 16 + TPB - 1) / TPB);

  WPtrs wp;
  wp.w[0] = lin_w;   wp.w[1] = lin_src_w; wp.w[2] = lin_dst_w;
  wp.w[3] = pos_w2;  wp.w[4] = attn_w1;   wp.w[5] = attn_w2;
  wp.w[6] = up_w;
  convert_weights_kernel<<<(7 * 4096 + TPB - 1) / TPB, TPB, 0, stream>>>(wp, wbf);

  node_proj_kernel<<<nodeBlocks, TPB, 0, stream>>>(x, wbf, val, asrc, adst, (int)N);

  moments_kernel<<<edgeBlocks, TPB, 0, stream>>>(pos, ei, E, M, mom);
  bn1_finalize_kernel<<<1, 64, 0, stream>>>(mom, pos_w1, pos_b1, pos_g1, pos_be1,
                                            invM, sab + 0);

  edge_gemm1_kernel<<<gemmBlocks, TPB, 0, stream>>>(pos, ei, E, M, pos_w1, pos_b1,
                                                    sab + 0, wbf + 3 * 4096, pos_b2,
                                                    hbuf, stats2);
  bn_finalize_kernel<<<1, 64, 0, stream>>>(stats2, pos_g2, pos_be2, invM, sab + 128);

  edge_attn_in_kernel<<<gemmBlocks, TPB, 0, stream>>>(hbuf, dbuf, M, E, ei,
                                                      sab + 128, asrc, adst,
                                                      wbf + 4 * 4096, attn_b1,
                                                      stats3);
  bn_finalize_kernel<<<1, 64, 0, stream>>>(stats3, attn_g1, attn_be1, invM,
                                           sab + 256);

  edge_gemm_kernel<<<gemmBlocks, TPB, 0, stream>>>(hbuf, M, sab + 256,
                                                   wbf + 5 * 4096, attn_b2, stats4);
  bn_finalize_kernel<<<1, 64, 0, stream>>>(stats4, attn_g2, attn_be2, invM,
                                           sab + 384);

  softmax_max_kernel<<<elemBlocks, TPB, 0, stream>>>(hbuf, M, E, ei, sab + 384,
                                                     segmax);
  softmax_exp_kernel<<<elemBlocks, TPB, 0, stream>>>(hbuf, M, E, ei, segmax,
                                                     segsum);
  scatter_kernel<<<elemBlocks, TPB, 0, stream>>>(hbuf, dbuf, M, E, ei, segsum, val,
                                                 outacc);

  final_kernel<<<nodeBlocks, TPB, 0, stream>>>(outacc, x, wbf + 6 * 4096, up_b,
                                               (float*)d_out, (int)N);
}
